// AdvancedMemorySystem_3478923510332
// MI455X (gfx1250) — compile-verified, matrix operand order checked by reading, not run on hardware
//
#include <hip/hip_runtime.h>

// MI455X / gfx1250 fused "episodic memory", round 3.
//   logits = x@W_epi^T + b_epi -> online softmax -> episodic = P@epi_mem
//   out = [episodic, x] @ W_cons^T + b_cons     (semantic branch dead -> skipped)
// Round-3: split-K over E (flash-decoding style), 4 splits -> 256 WGs for the
// heavy kernel; partial (m, l, unnormalized episodic) in ws; combine + consolidation
// GEMM in a second kernel. Kernel-1 LDS ~142KB -> 2 WGs/WGP can co-reside.

typedef __attribute__((ext_vector_type(16))) _Float16 v16h;
typedef __attribute__((ext_vector_type(8)))  _Float16 v8h;
typedef __attribute__((ext_vector_type(4)))  _Float16 v4h;
typedef __attribute__((ext_vector_type(8)))  float    v8f;

constexpr int kH     = 1024;
constexpr int kE     = 50000;
constexpr int kBT    = 2048;
constexpr int kM     = 32;                       // rows per workgroup
constexpr int kTILES = kBT / kM;                 // 64 row tiles
constexpr int kCHUNK = 128;                      // E-chunk
constexpr int kNCH   = (kE + kCHUNK - 1) / kCHUNK;   // 391
constexpr int kNW    = 8;                        // wave32s per workgroup
constexpr int kSPLIT = 4;                        // E-split factor
constexpr int kCPS   = (kNCH + kSPLIT - 1) / kSPLIT; // chunks per split (98)

static __device__ __forceinline__ v16h mk16(v8h lo, v8h hi) {
  return __builtin_shufflevector(lo, hi, 0,1,2,3,4,5,6,7,8,9,10,11,12,13,14,15);
}
static __device__ __forceinline__ v8h load8(const float* p) {
  const float4* q = reinterpret_cast<const float4*>(p);
  float4 f0 = q[0], f1 = q[1];
  v8h r;
  r[0]=(_Float16)f0.x; r[1]=(_Float16)f0.y; r[2]=(_Float16)f0.z; r[3]=(_Float16)f0.w;
  r[4]=(_Float16)f1.x; r[5]=(_Float16)f1.y; r[6]=(_Float16)f1.z; r[7]=(_Float16)f1.w;
  return r;
}
static __device__ __forceinline__ v8h load8(const _Float16* p) {
  return *reinterpret_cast<const v8h*>(p);
}
static __device__ __forceinline__ v8f wmma16(v16h a, v16h b, v8f c) {
  return __builtin_amdgcn_wmma_f32_16x16x32_f16(false, a, false, b, (short)0, c, false, false);
}

// ---------------- one-shot fp32 -> f16 conversion into d_ws
__global__ void cvt_f32_f16_kernel(const float* __restrict__ src,
                                   _Float16* __restrict__ dst, long n) {
  const long stride = (long)gridDim.x * blockDim.x * 4;
  for (long i = ((long)blockIdx.x * blockDim.x + threadIdx.x) * 4; i < n; i += stride) {
    float4 f = *reinterpret_cast<const float4*>(src + i);
    v4h h;
    h[0] = (_Float16)f.x; h[1] = (_Float16)f.y;
    h[2] = (_Float16)f.z; h[3] = (_Float16)f.w;
    *reinterpret_cast<v4h*>(dst + i) = h;
  }
}

// =====================================================================
// Kernel 1 (f16 split path): streams one E-range, emits partial softmax state.
// grid = (kTILES, kSPLIT), block = 256
// =====================================================================
__global__ __launch_bounds__(256)
void episodic_split_kernel(const float*    __restrict__ x,
                           const _Float16* __restrict__ W_epi,
                           const float*    __restrict__ b_epi,
                           const _Float16* __restrict__ epi_mem,
                           float* __restrict__ partAcc,   // [tile][split][kM][kH]
                           float* __restrict__ partML) {  // [tile][split][kM][2]
  __shared__ __align__(16) _Float16 sX[kM][kH];                  // 64 KB
  __shared__ __align__(16) _Float16 sP[kM][kCHUNK];              // 8 KB
  __shared__ __align__(16) _Float16 sBT[2][kCHUNK][kCHUNK + 8];  // 68 KB
  __shared__ float sRedMax[kNW][kM];
  __shared__ float sRedSum[kNW][kM];
  __shared__ float sRowM[kM];
  __shared__ float sRowScale[kM];
  __shared__ float sRowL[kM];

  const int tid  = threadIdx.x;
  const int w    = tid >> 5;
  const int lane = tid & 31;
  const int mn   = lane & 15;
  const int sub  = lane >> 4;
  const int tile  = blockIdx.x;
  const int split = blockIdx.y;
  const int rowbase = tile * kM;

  for (int idx = tid; idx < kM * (kH / 4); idx += 256) {
    const int r  = idx >> 8;
    const int c4 = (idx & 255) << 2;
    float4 f = *reinterpret_cast<const float4*>(x + (long)(rowbase + r) * kH + c4);
    sX[r][c4 + 0] = (_Float16)f.x; sX[r][c4 + 1] = (_Float16)f.y;
    sX[r][c4 + 2] = (_Float16)f.z; sX[r][c4 + 3] = (_Float16)f.w;
  }
  if (tid < kM) {
    sRowM[tid] = -__builtin_inff();
    sRowL[tid] = 0.0f;
    sRowScale[tid] = 0.0f;
  }
  __syncthreads();

  v8f zero8 = {};
  v8f acc[2][8];
  #pragma unroll
  for (int r = 0; r < 2; ++r)
    #pragma unroll
    for (int p = 0; p < 8; ++p) acc[r][p] = zero8;

  auto stage_panel = [&](int e0, int p, int buf) {
    for (int idx = tid; idx < kCHUNK * 16; idx += 256) {
      const int  er = idx >> 4;
      const int  h8 = (idx & 15) << 3;
      const long e  = (long)e0 + er;
      v8h v = {};
      if (e < kE) v = load8(epi_mem + e * kH + p * 128 + h8);
      #pragma unroll
      for (int j = 0; j < 8; ++j) sBT[buf][h8 + j][er] = v[j];
    }
  };

  const int chBeg = split * kCPS;
  const int chEnd = (chBeg + kCPS < kNCH) ? (chBeg + kCPS) : kNCH;

  for (int ch = chBeg; ch < chEnd; ++ch) {
    const int  e0     = ch * kCHUNK;
    const bool wvalid = (e0 + w * 16) < kE;
    const int  ecol   = e0 + w * 16 + mn;

    // -------- phase 1: two 16x16 logit tiles per wave, shared B operand
    v8f cl[2] = {zero8, zero8};
    if (wvalid) {
      const _Float16* Wr = W_epi + (long)ecol * kH;
      if (ecol + kCHUNK < kE)
        __builtin_prefetch(W_epi + (long)(ecol + kCHUNK) * kH, 0, 1);
      #pragma unroll 4
      for (int kk = 0; kk < 32; ++kk) {
        const int kb = kk * 32 + sub * 8;
        v16h b = mk16(load8(Wr + kb), load8(Wr + kb + 16));
        #pragma unroll
        for (int r = 0; r < 2; ++r) {
          v16h a = mk16(*reinterpret_cast<const v8h*>(&sX[r * 16 + mn][kb]),
                        *reinterpret_cast<const v8h*>(&sX[r * 16 + mn][kb + 16]));
          cl[r] = wmma16(a, b, cl[r]);
        }
      }
      const float be = b_epi[ecol];
      #pragma unroll
      for (int r = 0; r < 2; ++r)
        #pragma unroll
        for (int i = 0; i < 8; ++i) cl[r][i] += be;
    }

    // -------- online softmax bookkeeping
    float rmax[2][8];
    #pragma unroll
    for (int r = 0; r < 2; ++r)
      #pragma unroll
      for (int i = 0; i < 8; ++i) {
        float v = wvalid ? cl[r][i] : -__builtin_inff();
        #pragma unroll
        for (int off = 1; off < 16; off <<= 1) v = fmaxf(v, __shfl_xor(v, off));
        rmax[r][i] = v;
      }
    if (lane == 0) {
      #pragma unroll
      for (int r = 0; r < 2; ++r)
        #pragma unroll
        for (int i = 0; i < 8; ++i) sRedMax[w][r * 16 + i] = rmax[r][i];
    }
    if (lane == 16) {
      #pragma unroll
      for (int r = 0; r < 2; ++r)
        #pragma unroll
        for (int i = 0; i < 8; ++i) sRedMax[w][r * 16 + 8 + i] = rmax[r][i];
    }
    __syncthreads();

    if (tid < kM) {
      float cm = -__builtin_inff();
      #pragma unroll
      for (int ww = 0; ww < kNW; ++ww) cm = fmaxf(cm, sRedMax[ww][tid]);
      const float mprev = sRowM[tid];
      const float mnew  = fmaxf(mprev, cm);
      sRowScale[tid] = (mprev == -__builtin_inff()) ? 0.0f : __expf(mprev - mnew);
      sRowM[tid] = mnew;
    }
    __syncthreads();

    float ps[2][8], sc[2][8];
    #pragma unroll
    for (int r = 0; r < 2; ++r)
      #pragma unroll
      for (int i = 0; i < 8; ++i) {
        const int row = r * 16 + i + 8 * sub;
        sc[r][i] = sRowScale[row];
        ps[r][i] = wvalid ? __expf(cl[r][i] - sRowM[row]) : 0.0f;
      }
    #pragma unroll
    for (int r = 0; r < 2; ++r)
      #pragma unroll
      for (int i = 0; i < 8; ++i)
        sP[r * 16 + i + 8 * sub][w * 16 + mn] = (_Float16)ps[r][i];
    #pragma unroll
    for (int r = 0; r < 2; ++r)
      #pragma unroll
      for (int p = 0; p < 8; ++p)
        #pragma unroll
        for (int i = 0; i < 8; ++i) acc[r][p][i] *= sc[r][i];

    float rsum[2][8];
    #pragma unroll
    for (int r = 0; r < 2; ++r)
      #pragma unroll
      for (int i = 0; i < 8; ++i) {
        float v = ps[r][i];
        #pragma unroll
        for (int off = 1; off < 16; off <<= 1) v += __shfl_xor(v, off);
        rsum[r][i] = v;
      }
    if (lane == 0) {
      #pragma unroll
      for (int r = 0; r < 2; ++r)
        #pragma unroll
        for (int i = 0; i < 8; ++i) sRedSum[w][r * 16 + i] = rsum[r][i];
    }
    if (lane == 16) {
      #pragma unroll
      for (int r = 0; r < 2; ++r)
        #pragma unroll
        for (int i = 0; i < 8; ++i) sRedSum[w][r * 16 + 8 + i] = rsum[r][i];
    }

    stage_panel(e0, 0, 0);     // overlap panel-0 staging with softmax tail
    __syncthreads();

    if (tid < kM) {
      float s = 0.0f;
      #pragma unroll
      for (int ww = 0; ww < kNW; ++ww) s += sRedSum[ww][tid];
      sRowL[tid] = sRowL[tid] * sRowScale[tid] + s;
    }

    // -------- phase 2: episodic += P @ epi_mem[chunk] (double-buffered panels)
    for (int p = 0; p < 8; ++p) {
      if (p < 7) stage_panel(e0, p + 1, (p + 1) & 1);
      const int buf = p & 1;
      #pragma unroll
      for (int kk = 0; kk < 4; ++kk) {
        const int kb = kk * 32 + sub * 8;
        v16h b = mk16(*reinterpret_cast<const v8h*>(&sBT[buf][w * 16 + mn][kb]),
                      *reinterpret_cast<const v8h*>(&sBT[buf][w * 16 + mn][kb + 16]));
        #pragma unroll
        for (int r = 0; r < 2; ++r) {
          v16h a = mk16(*reinterpret_cast<const v8h*>(&sP[r * 16 + mn][kb]),
                        *reinterpret_cast<const v8h*>(&sP[r * 16 + mn][kb + 16]));
          acc[r][p] = wmma16(a, b, acc[r][p]);
        }
      }
      __syncthreads();
    }
  }

  // -------- emit partial state
  float* accOut = partAcc + (long)(tile * kSPLIT + split) * kM * kH;
  #pragma unroll
  for (int r = 0; r < 2; ++r)
    #pragma unroll
    for (int p = 0; p < 8; ++p)
      #pragma unroll
      for (int i = 0; i < 8; ++i)
        accOut[(long)(r * 16 + i + 8 * sub) * kH + p * 128 + w * 16 + mn] = acc[r][p][i];
  if (tid < kM) {
    float* ml = partML + (long)(tile * kSPLIT + split) * kM * 2;
    ml[tid * 2 + 0] = sRowM[tid];
    ml[tid * 2 + 1] = sRowL[tid];
  }
}

// =====================================================================
// Kernel 2: combine partials -> episodic (f16 in LDS) -> consolidation GEMM.
// grid = (kTILES, 2 column halves), block = 256
// =====================================================================
__global__ __launch_bounds__(256)
void combine_cons_kernel(const float*    __restrict__ x,
                         const float*    __restrict__ partAcc,
                         const float*    __restrict__ partML,
                         const _Float16* __restrict__ W_cons,
                         const float*    __restrict__ b_cons,
                         float* __restrict__ out) {
  __shared__ __align__(16) _Float16 sEpi[kM][kH];   // 64 KB
  __shared__ __align__(16) _Float16 sX[kM][kH];     // 64 KB
  __shared__ float sCoeff[kSPLIT][kM];
  __shared__ float sInvL[kM];

  const int tid  = threadIdx.x;
  const int w    = tid >> 5;
  const int lane = tid & 31;
  const int mn   = lane & 15;
  const int sub  = lane >> 4;
  const int tile = blockIdx.x;
  const int cs   = blockIdx.y;          // output column half
  const int rowbase = tile * kM;

  if (tid < kM) {
    float m[kSPLIT], l[kSPLIT];
    #pragma unroll
    for (int s = 0; s < kSPLIT; ++s) {
      const long base = ((long)(tile * kSPLIT + s) * kM + tid) * 2;
      m[s] = partML[base];
      l[s] = partML[base + 1];
    }
    float gm = m[0];
    #pragma unroll
    for (int s = 1; s < kSPLIT; ++s) gm = fmaxf(gm, m[s]);
    float L = 0.0f;
    #pragma unroll
    for (int s = 0; s < kSPLIT; ++s) {
      const float c = __expf(m[s] - gm);
      sCoeff[s][tid] = c;
      L += c * l[s];
    }
    sInvL[tid] = 1.0f / L;
  }
  for (int idx = tid; idx < kM * (kH / 4); idx += 256) {
    const int r  = idx >> 8;
    const int c4 = (idx & 255) << 2;
    float4 f = *reinterpret_cast<const float4*>(x + (long)(rowbase + r) * kH + c4);
    sX[r][c4 + 0] = (_Float16)f.x; sX[r][c4 + 1] = (_Float16)f.y;
    sX[r][c4 + 2] = (_Float16)f.z; sX[r][c4 + 3] = (_Float16)f.w;
  }
  __syncthreads();

  for (int idx = tid; idx < kM * (kH / 4); idx += 256) {
    const int row = idx >> 8;
    const int c4  = (idx & 255) << 2;
    float ax = 0.f, ay = 0.f, az = 0.f, aw = 0.f;
    #pragma unroll
    for (int s = 0; s < kSPLIT; ++s) {
      const float4 v = *reinterpret_cast<const float4*>(
          partAcc + ((long)(tile * kSPLIT + s) * kM + row) * kH + c4);
      const float c = sCoeff[s][row];
      ax += c * v.x; ay += c * v.y; az += c * v.z; aw += c * v.w;
    }
    const float inv = sInvL[row];
    sEpi[row][c4 + 0] = (_Float16)(ax * inv);
    sEpi[row][c4 + 1] = (_Float16)(ay * inv);
    sEpi[row][c4 + 2] = (_Float16)(az * inv);
    sEpi[row][c4 + 3] = (_Float16)(aw * inv);
  }
  __syncthreads();

  // out = [episodic, x] @ W_cons^T + b_cons ; this WG covers 512 columns
  v8f zero8 = {};
  v8f o[2][4];
  #pragma unroll
  for (int r = 0; r < 2; ++r)
    #pragma unroll
    for (int t = 0; t < 4; ++t) o[r][t] = zero8;
  for (int kk = 0; kk < 64; ++kk) {
    v16h a[2];
    #pragma unroll
    for (int r = 0; r < 2; ++r) {
      const _Float16* ar = (kk < 32) ? &sEpi[r * 16 + mn][kk * 32]
                                     : &sX[r * 16 + mn][(kk - 32) * 32];
      a[r] = mk16(*reinterpret_cast<const v8h*>(ar + sub * 8),
                  *reinterpret_cast<const v8h*>(ar + 16 + sub * 8));
    }
    #pragma unroll
    for (int t = 0; t < 4; ++t) {
      const int ncol = cs * 512 + w * 64 + t * 16 + mn;
      const _Float16* Wr = W_cons + (long)ncol * (2 * kH) + kk * 32;
      v16h b = mk16(load8(Wr + sub * 8), load8(Wr + 16 + sub * 8));
      #pragma unroll
      for (int r = 0; r < 2; ++r) o[r][t] = wmma16(a[r], b, o[r][t]);
    }
  }
  #pragma unroll
  for (int t = 0; t < 4; ++t) {
    const int   ncol = cs * 512 + w * 64 + t * 16 + mn;
    const float bias = b_cons[ncol];
    #pragma unroll
    for (int r = 0; r < 2; ++r)
      #pragma unroll
      for (int i = 0; i < 8; ++i)
        out[(long)(rowbase + r * 16 + i + 8 * sub) * kH + ncol] = o[r][t][i] + bias;
  }
}

// =====================================================================
// Monolithic fp32 fallback (only used when workspace is too small)
// =====================================================================
template <typename WT>
__global__ __launch_bounds__(256)
void fused_mono_kernel(const float* __restrict__ x,
                       const WT*    __restrict__ W_epi,
                       const float* __restrict__ b_epi,
                       const WT*    __restrict__ epi_mem,
                       const WT*    __restrict__ W_cons,
                       const float* __restrict__ b_cons,
                       float* __restrict__ out) {
  __shared__ __align__(16) _Float16 sX[kM][kH];
  __shared__ __align__(16) _Float16 sEpi[kM][kH];
  __shared__ __align__(16) _Float16 sP[kM][kCHUNK];
  __shared__ __align__(16) _Float16 sBT[2][kCHUNK][kCHUNK + 8];
  __shared__ float sRedMax[kNW][kM];
  __shared__ float sRedSum[kNW][kM];
  __shared__ float sRowM[kM];
  __shared__ float sRowScale[kM];
  __shared__ float sRowL[kM];

  const int tid  = threadIdx.x;
  const int w    = tid >> 5;
  const int lane = tid & 31;
  const int mn   = lane & 15;
  const int sub  = lane >> 4;
  const int rowbase = blockIdx.x * kM;

  for (int idx = tid; idx < kM * (kH / 4); idx += 256) {
    const int r  = idx >> 8;
    const int c4 = (idx & 255) << 2;
    float4 f = *reinterpret_cast<const float4*>(x + (long)(rowbase + r) * kH + c4);
    sX[r][c4 + 0] = (_Float16)f.x; sX[r][c4 + 1] = (_Float16)f.y;
    sX[r][c4 + 2] = (_Float16)f.z; sX[r][c4 + 3] = (_Float16)f.w;
  }
  if (tid < kM) {
    sRowM[tid] = -__builtin_inff();
    sRowL[tid] = 0.0f;
    sRowScale[tid] = 0.0f;
  }
  __syncthreads();

  v8f zero8 = {};
  v8f acc[2][8];
  #pragma unroll
  for (int r = 0; r < 2; ++r)
    #pragma unroll
    for (int p = 0; p < 8; ++p) acc[r][p] = zero8;

  auto stage_panel = [&](int e0, int p, int buf) {
    for (int idx = tid; idx < kCHUNK * 16; idx += 256) {
      const int  er = idx >> 4;
      const int  h8 = (idx & 15) << 3;
      const long e  = (long)e0 + er;
      v8h v = {};
      if (e < kE) v = load8(epi_mem + e * kH + p * 128 + h8);
      #pragma unroll
      for (int j = 0; j < 8; ++j) sBT[buf][h8 + j][er] = v[j];
    }
  };

  for (int ch = 0; ch < kNCH; ++ch) {
    const int  e0     = ch * kCHUNK;
    const bool wvalid = (e0 + w * 16) < kE;
    const int  ecol   = e0 + w * 16 + mn;

    v8f cl[2] = {zero8, zero8};
    if (wvalid) {
      const WT* Wr = W_epi + (long)ecol * kH;
      #pragma unroll 4
      for (int kk = 0; kk < 32; ++kk) {
        const int kb = kk * 32 + sub * 8;
        v16h b = mk16(load8(Wr + kb), load8(Wr + kb + 16));
        #pragma unroll
        for (int r = 0; r < 2; ++r) {
          v16h a = mk16(*reinterpret_cast<const v8h*>(&sX[r * 16 + mn][kb]),
                        *reinterpret_cast<const v8h*>(&sX[r * 16 + mn][kb + 16]));
          cl[r] = wmma16(a, b, cl[r]);
        }
      }
      const float be = b_epi[ecol];
      #pragma unroll
      for (int r = 0; r < 2; ++r)
        #pragma unroll
        for (int i = 0; i < 8; ++i) cl[r][i] += be;
    }

    float rmax[2][8];
    #pragma unroll
    for (int r = 0; r < 2; ++r)
      #pragma unroll
      for (int i = 0; i < 8; ++i) {
        float v = wvalid ? cl[r][i] : -__builtin_inff();
        #pragma unroll
        for (int off = 1; off < 16; off <<= 1) v = fmaxf(v, __shfl_xor(v, off));
        rmax[r][i] = v;
      }
    if (lane == 0) {
      #pragma unroll
      for (int r = 0; r < 2; ++r)
        #pragma unroll
        for (int i = 0; i < 8; ++i) sRedMax[w][r * 16 + i] = rmax[r][i];
    }
    if (lane == 16) {
      #pragma unroll
      for (int r = 0; r < 2; ++r)
        #pragma unroll
        for (int i = 0; i < 8; ++i) sRedMax[w][r * 16 + 8 + i] = rmax[r][i];
    }
    __syncthreads();

    if (tid < kM) {
      float cm = -__builtin_inff();
      #pragma unroll
      for (int ww = 0; ww < kNW; ++ww) cm = fmaxf(cm, sRedMax[ww][tid]);
      const float mprev = sRowM[tid];
      const float mnew  = fmaxf(mprev, cm);
      sRowScale[tid] = (mprev == -__builtin_inff()) ? 0.0f : __expf(mprev - mnew);
      sRowM[tid] = mnew;
    }
    __syncthreads();

    float ps[2][8], sc[2][8];
    #pragma unroll
    for (int r = 0; r < 2; ++r)
      #pragma unroll
      for (int i = 0; i < 8; ++i) {
        const int row = r * 16 + i + 8 * sub;
        sc[r][i] = sRowScale[row];
        ps[r][i] = wvalid ? __expf(cl[r][i] - sRowM[row]) : 0.0f;
      }
    #pragma unroll
    for (int r = 0; r < 2; ++r)
      #pragma unroll
      for (int i = 0; i < 8; ++i)
        sP[r * 16 + i + 8 * sub][w * 16 + mn] = (_Float16)ps[r][i];
    #pragma unroll
    for (int r = 0; r < 2; ++r)
      #pragma unroll
      for (int p = 0; p < 8; ++p)
        #pragma unroll
        for (int i = 0; i < 8; ++i) acc[r][p][i] *= sc[r][i];

    float rsum[2][8];
    #pragma unroll
    for (int r = 0; r < 2; ++r)
      #pragma unroll
      for (int i = 0; i < 8; ++i) {
        float v = ps[r][i];
        #pragma unroll
        for (int off = 1; off < 16; off <<= 1) v += __shfl_xor(v, off);
        rsum[r][i] = v;
      }
    if (lane == 0) {
      #pragma unroll
      for (int r = 0; r < 2; ++r)
        #pragma unroll
        for (int i = 0; i < 8; ++i) sRedSum[w][r * 16 + i] = rsum[r][i];
    }
    if (lane == 16) {
      #pragma unroll
      for (int r = 0; r < 2; ++r)
        #pragma unroll
        for (int i = 0; i < 8; ++i) sRedSum[w][r * 16 + 8 + i] = rsum[r][i];
    }

    stage_panel(e0, 0, 0);
    __syncthreads();

    if (tid < kM) {
      float s = 0.0f;
      #pragma unroll
      for (int ww = 0; ww < kNW; ++ww) s += sRedSum[ww][tid];
      sRowL[tid] = sRowL[tid] * sRowScale[tid] + s;
    }

    for (int p = 0; p < 8; ++p) {
      if (p < 7) stage_panel(e0, p + 1, (p + 1) & 1);
      const int buf = p & 1;
      #pragma unroll
      for (int kk = 0; kk < 4; ++kk) {
        const int kb = kk * 32 + sub * 8;
        v16h b = mk16(*reinterpret_cast<const v8h*>(&sBT[buf][w * 16 + mn][kb]),
                      *reinterpret_cast<const v8h*>(&sBT[buf][w * 16 + mn][kb + 16]));
        #pragma unroll
        for (int r = 0; r < 2; ++r) {
          v16h a = mk16(*reinterpret_cast<const v8h*>(&sP[r * 16 + mn][kb]),
                        *reinterpret_cast<const v8h*>(&sP[r * 16 + mn][kb + 16]));
          acc[r][p] = wmma16(a, b, acc[r][p]);
        }
      }
      __syncthreads();
    }
  }

  {
    float invl[2][8];
    #pragma unroll
    for (int r = 0; r < 2; ++r)
      #pragma unroll
      for (int i = 0; i < 8; ++i) invl[r][i] = 1.0f / sRowL[r * 16 + i + 8 * sub];
    #pragma unroll
    for (int r = 0; r < 2; ++r)
      #pragma unroll
      for (int p = 0; p < 8; ++p)
        #pragma unroll
        for (int i = 0; i < 8; ++i)
          sEpi[r * 16 + i + 8 * sub][p * 128 + w * 16 + mn] =
              (_Float16)(acc[r][p][i] * invl[r][i]);
  }
  __syncthreads();

  v8f o[2][8];
  #pragma unroll
  for (int r = 0; r < 2; ++r)
    #pragma unroll
    for (int t = 0; t < 8; ++t) o[r][t] = zero8;
  for (int kk = 0; kk < 64; ++kk) {
    v16h a[2];
    #pragma unroll
    for (int r = 0; r < 2; ++r) {
      const _Float16* ar = (kk < 32) ? &sEpi[r * 16 + mn][kk * 32]
                                     : &sX[r * 16 + mn][(kk - 32) * 32];
      a[r] = mk16(*reinterpret_cast<const v8h*>(ar + sub * 8),
                  *reinterpret_cast<const v8h*>(ar + 16 + sub * 8));
    }
    #pragma unroll
    for (int t = 0; t < 8; ++t) {
      const int ncol = w * 128 + t * 16 + mn;
      const WT* Wr = W_cons + (long)ncol * (2 * kH) + kk * 32;
      v16h b = mk16(load8(Wr + sub * 8), load8(Wr + 16 + sub * 8));
      #pragma unroll
      for (int r = 0; r < 2; ++r) o[r][t] = wmma16(a[r], b, o[r][t]);
    }
  }
  #pragma unroll
  for (int t = 0; t < 8; ++t) {
    const int   ncol = w * 128 + t * 16 + mn;
    const float bias = b_cons[ncol];
    #pragma unroll
    for (int r = 0; r < 2; ++r)
      #pragma unroll
      for (int i = 0; i < 8; ++i)
        out[(long)(rowbase + r * 16 + i + 8 * sub) * kH + ncol] = o[r][t][i] + bias;
  }
}

// =====================================================================
extern "C" void kernel_launch(void* const* d_in, const int* in_sizes, int n_in,
                              void* d_out, int out_size, void* d_ws, size_t ws_size,
                              hipStream_t stream) {
  (void)in_sizes; (void)n_in; (void)out_size;
  const float* x       = (const float*)d_in[0];
  const float* W_epi   = (const float*)d_in[1];
  const float* b_epi   = (const float*)d_in[2];
  const float* epi_mem = (const float*)d_in[3];
  // d_in[4..6] = semantic branch, dead in the reference output
  const float* W_cons  = (const float*)d_in[7];
  const float* b_cons  = (const float*)d_in[8];
  float* out = (float*)d_out;

  const long nW = (long)kE * kH;
  const long nC = (long)2 * kH * kH;
  const size_t bytes_f16  = (size_t)(2 * nW + nC) * sizeof(_Float16);   // ~209 MB
  const size_t bytes_part = (size_t)kTILES * kSPLIT * kM * kH * sizeof(float)
                          + (size_t)kTILES * kSPLIT * kM * 2 * sizeof(float); // ~33.6 MB
  const size_t need = bytes_f16 + bytes_part;

  if (ws_size >= need) {
    _Float16* wE = (_Float16*)d_ws;
    _Float16* wM = wE + nW;
    _Float16* wC = wM + nW;
    float* partAcc = (float*)((char*)d_ws + bytes_f16);
    float* partML  = partAcc + (size_t)kTILES * kSPLIT * kM * kH;

    cvt_f32_f16_kernel<<<4096, 256, 0, stream>>>(W_epi,   wE, nW);
    cvt_f32_f16_kernel<<<4096, 256, 0, stream>>>(epi_mem, wM, nW);
    cvt_f32_f16_kernel<<<64,   256, 0, stream>>>(W_cons,  wC, nC);

    episodic_split_kernel<<<dim3(kTILES, kSPLIT), 256, 0, stream>>>(
        x, wE, b_epi, wM, partAcc, partML);
    combine_cons_kernel<<<dim3(kTILES, 2), 256, 0, stream>>>(
        x, partAcc, partML, wC, b_cons, out);
  } else {
    fused_mono_kernel<float><<<dim3(kTILES), 256, 0, stream>>>(
        x, W_epi, b_epi, epi_mem, W_cons, b_cons, out);
  }
}
